// GCN_71597104824805
// MI455X (gfx1250) — matrix operand order repaired and checked by reading, not compiled, not run on hardware
//
#include <hip/hip_runtime.h>
#include <cstdint>
#include <cstddef>

typedef __attribute__((ext_vector_type(16))) __bf16 v16bf;
typedef __attribute__((ext_vector_type(8)))  __bf16 v8bf;
typedef __attribute__((ext_vector_type(8)))  float  v8f;
typedef __attribute__((ext_vector_type(4)))  int    v4i;

#define GCN_NODES 50000
#define GCN_EDGES 600000

#define LDS_AS    __attribute__((address_space(3)))
#define GLOBAL_AS __attribute__((address_space(1)))

typedef GLOBAL_AS v4i* gv4i_p;
typedef LDS_AS    v4i* lv4i_p;

#if defined(__AMDGCN__) && __has_builtin(__builtin_amdgcn_global_load_async_to_lds_b128) && \
    __has_builtin(__builtin_amdgcn_s_wait_asynccnt)
#define USE_ASYNC_LDS 1
#else
#define USE_ASYNC_LDS 0
#endif

__device__ __forceinline__ __bf16 f2bf(float f) {
  unsigned u = __builtin_bit_cast(unsigned, f);
  unsigned r = u + 0x7FFFu + ((u >> 16) & 1u);     // round-to-nearest-even
  unsigned short s = (unsigned short)(r >> 16);
  return __builtin_bit_cast(__bf16, s);
}

__global__ void zero_f32(float* __restrict__ p, int n) {
  for (int i = blockIdx.x * blockDim.x + threadIdx.x; i < n; i += gridDim.x * blockDim.x)
    p[i] = 0.0f;
}

__global__ void degree_kernel(const int* __restrict__ src, const int* __restrict__ dst,
                              float* __restrict__ deg_out, float* __restrict__ deg_in, int nE) {
  int e = blockIdx.x * blockDim.x + threadIdx.x;
  if (e >= nE) return;
  atomicAdd(&deg_out[src[e]], 1.0f);
  atomicAdd(&deg_in[dst[e]], 1.0f);
}

__global__ void norm_kernel(float* __restrict__ deg, int n) {
  int i = blockIdx.x * blockDim.x + threadIdx.x;
  if (i >= n) return;
  float d = deg[i];
  deg[i] = (d > 0.0f) ? rsqrtf(fmaxf(d, 1.0f)) : 0.0f;
}

// Repack row-major W[128][Nout] (fp32) into per-lane WMMA B-fragment order (bf16).
// Packed element idx = (((kt*ntiles + nt)*32 + lane)*16 + i)
//   maps to W[kt*32 + (lane>>4)*16 + i][nt*16 + (lane&15)].
__global__ void packW_kernel(const float* __restrict__ W, __bf16* __restrict__ Wp, int Nout) {
  const int ntiles = Nout >> 4;
  const int total  = 128 * Nout;
  for (int idx = blockIdx.x * blockDim.x + threadIdx.x; idx < total;
       idx += gridDim.x * blockDim.x) {
    int i    = idx & 15;
    int lane = (idx >> 4) & 31;
    int rest = idx >> 9;
    int nt   = rest % ntiles;
    int kt   = rest / ntiles;
    int k    = kt * 32 + (lane >> 4) * 16 + i;
    int c    = nt * 16 + (lane & 15);
    Wp[idx]  = f2bf(W[k * Nout + c]);
  }
}

// One wave per edge: gather X[src]*norm_src, atomic-scatter into Y[dst] (128 feats).
__global__ void spmm128_kernel(const float* __restrict__ X, const float* __restrict__ norm_src,
                               const int* __restrict__ src, const int* __restrict__ dst,
                               float* __restrict__ Y, int nE) {
  int e = blockIdx.x * (blockDim.x >> 5) + (threadIdx.x >> 5);
  if (e >= nE) return;
  int lane = threadIdx.x & 31;
  int s = src[e], d = dst[e];
  float w = norm_src[s];
  const float4 xv = *(const float4*)(X + (size_t)s * 128 + lane * 4);
  float* yp = Y + (size_t)d * 128 + lane * 4;
  atomicAdd(yp + 0, xv.x * w);
  atomicAdd(yp + 1, xv.y * w);
  atomicAdd(yp + 2, xv.z * w);
  atomicAdd(yp + 3, xv.w * w);
}

// One thread per (edge, feature) for the 16-wide final layer aggregation.
__global__ void spmm16_kernel(const float* __restrict__ H, const int* __restrict__ src,
                              const int* __restrict__ dst, float* __restrict__ agg, int nE) {
  int idx = blockIdx.x * blockDim.x + threadIdx.x;
  if (idx >= nE * 16) return;
  int e = idx >> 4, k = idx & 15;
  atomicAdd(&agg[(size_t)dst[e] * 16 + k], H[(size_t)src[e] * 16 + k]);
}

// Y[16-row tile] = act( (X * rowscale) @ W + bias ).  K fixed at 128.
// Block = 32*NTILES threads; wave nt computes the 16x16 tile for columns nt*16..+15.
// Grid must tile rows exactly (50000 = 3125 * 16) -> no bounds checks.
template <int NTILES, bool FUSE>
__global__ __launch_bounds__(32 * NTILES) void gemm_wmma_kernel(
    const float* __restrict__ X, const float* __restrict__ rowscale,
    const __bf16* __restrict__ Wp, const float* __restrict__ bias,
    float* __restrict__ Y) {
  constexpr int NOUT     = NTILES * 16;
  constexpr int NTHREADS = 32 * NTILES;
  constexpr int OSTRIDE  = NOUT + 4;               // padded f32 output stride (bank-safe)

  // sraw: raw fp32 tile (2048 floats) first, reused as padded output tile (16*OSTRIDE <= 2112)
  __shared__ __attribute__((aligned(32))) float  sraw[16 * 132];
  __shared__ __attribute__((aligned(32))) __bf16 sa[16 * 136];   // bf16 A tile, padded stride

  const int tid     = threadIdx.x;
  const int rowbase = blockIdx.x << 4;
  const float* Xtile = X + (size_t)rowbase * 128;  // 16 full rows => one contiguous 8KB span

#if USE_ASYNC_LDS
  // Async global->LDS copy of the contiguous fp32 tile (ASYNCcnt-tracked, no VGPR roundtrip).
  for (int i = tid; i < 512; i += NTHREADS) {      // 512 x 16B = 8KB
    unsigned long long gaddr =
        (unsigned long long)(const void*)Xtile + (unsigned long long)i * 16u;
    unsigned laddr =
        (unsigned)(unsigned long long)(const void*)sraw + (unsigned)i * 16u;
    __builtin_amdgcn_global_load_async_to_lds_b128((gv4i_p)gaddr, (lv4i_p)laddr, 0, 0);
  }
  __builtin_amdgcn_s_wait_asynccnt(0);
  __syncthreads();
  for (int idx = tid; idx < 2048; idx += NTHREADS) {
    int r = idx >> 7, c = idx & 127;
    sa[r * 136 + c] = f2bf(sraw[idx] * rowscale[rowbase + r]);
  }
#else
  for (int idx = tid; idx < 2048; idx += NTHREADS) {
    int r = idx >> 7, c = idx & 127;
    sa[r * 136 + c] = f2bf(Xtile[idx] * rowscale[rowbase + r]);
  }
#endif
  __syncthreads();

  const int lane = tid & 31;
  const int nt   = tid >> 5;
  const int rr   = lane & 15;   // A row within tile / C column within tile
  const int hv   = lane >> 4;   // lane-half selector

  v8f acc = {};
  const __bf16* bp = Wp + ((size_t)nt * 32 + lane) * 16;
  constexpr size_t BSTEP = (size_t)NTILES * 32 * 16;

#pragma unroll
  for (int kt = 0; kt < 4; ++kt) {
    // A fragment: K = kt*32 + hv*16 .. +15 of row rr.
    const __bf16* ap = sa + rr * 136 + kt * 32 + hv * 16;
    v8bf lo = *(const v8bf*)(ap);
    v8bf hi = *(const v8bf*)(ap + 8);
    v16bf a = __builtin_shufflevector(lo, hi, 0, 1, 2, 3, 4, 5, 6, 7,
                                              8, 9, 10, 11, 12, 13, 14, 15);
    v16bf b = *(const v16bf*)(bp + kt * BSTEP);
    acc = __builtin_amdgcn_wmma_f32_16x16x32_bf16(false, a, false, b,
                                                  (short)0, acc, false, false);
  }

  // Transpose accumulators through LDS so stores are row-major coalesced b128s.
  float* sout = sraw;           // safe: all sraw reads completed before prior barrier
  const int col = nt * 16 + rr;
#pragma unroll
  for (int v = 0; v < 8; ++v) { // C/D layout: VGPR v holds M = v + 8*(lane>=16)
    sout[(hv * 8 + v) * OSTRIDE + col] = acc[v];
  }
  __syncthreads();

  // Each thread writes 8 consecutive floats of one output row.
  const int idx = tid * 8;
  const int r = idx / NOUT;
  const int c = idx % NOUT;
  const float* sp = sout + r * OSTRIDE + c;
  float4 o0 = *(const float4*)(sp);
  float4 o1 = *(const float4*)(sp + 4);
  if constexpr (FUSE) {
    const float4 g0 = *(const float4*)(bias + c);
    const float4 g1 = *(const float4*)(bias + c + 4);
    o0.x = fmaxf(o0.x + g0.x, 0.0f); o0.y = fmaxf(o0.y + g0.y, 0.0f);
    o0.z = fmaxf(o0.z + g0.z, 0.0f); o0.w = fmaxf(o0.w + g0.w, 0.0f);
    o1.x = fmaxf(o1.x + g1.x, 0.0f); o1.y = fmaxf(o1.y + g1.y, 0.0f);
    o1.z = fmaxf(o1.z + g1.z, 0.0f); o1.w = fmaxf(o1.w + g1.w, 0.0f);
  }
  float* yp = Y + (size_t)(rowbase + r) * NOUT + c;
  *(float4*)(yp)     = o0;
  *(float4*)(yp + 4) = o1;
}

__global__ void final_kernel(const float* __restrict__ agg, const float* __restrict__ norm_dst,
                             const float* __restrict__ b2, float* __restrict__ out, int n) {
  int idx = blockIdx.x * blockDim.x + threadIdx.x;
  if (idx >= n * 16) return;
  int row = idx >> 4, c = idx & 15;
  float v = agg[idx] * norm_dst[row] + b2[c];
  out[idx] = 1.0f / (1.0f + __expf(-v)) + 1e-8f;
}

extern "C" void kernel_launch(void* const* d_in, const int* in_sizes, int n_in,
                              void* d_out, int out_size, void* d_ws, size_t ws_size,
                              hipStream_t stream) {
  (void)in_sizes; (void)n_in; (void)out_size; (void)ws_size;
  const float* features = (const float*)d_in[0];
  const int*   src      = (const int*)d_in[1];
  const int*   dst      = (const int*)d_in[2];
  const float* W0       = (const float*)d_in[3];
  const float* b0       = (const float*)d_in[4];
  const float* W1       = (const float*)d_in[5];
  const float* b1       = (const float*)d_in[6];
  const float* W2       = (const float*)d_in[7];
  const float* b2       = (const float*)d_in[8];
  float* out = (float*)d_out;

  // Workspace carve-up (fp32 region first, bf16 packed weights after; 32B-aligned).
  float* ws       = (float*)d_ws;
  float* norm_src = ws;                                   // 50000
  float* norm_dst = norm_src + GCN_NODES;                 // 50000
  float* bufA     = norm_dst + GCN_NODES;                 // 50000*128
  float* bufB     = bufA + (size_t)GCN_NODES * 128;       // 50000*128
  float* h2       = bufB + (size_t)GCN_NODES * 128;       // 50000*16
  float* agg2     = h2 + (size_t)GCN_NODES * 16;          // 50000*16
  __bf16* Wp0     = (__bf16*)(agg2 + (size_t)GCN_NODES * 16);
  __bf16* Wp1     = Wp0 + 128 * 128;
  __bf16* Wp2     = Wp1 + 128 * 128;

  const int nN = GCN_NODES, nE = GCN_EDGES;

  // Degrees -> norms
  zero_f32<<<(2 * nN + 255) / 256, 256, 0, stream>>>(norm_src, 2 * nN);
  degree_kernel<<<(nE + 255) / 256, 256, 0, stream>>>(src, dst, norm_src, norm_dst, nE);
  norm_kernel<<<(nN + 255) / 256, 256, 0, stream>>>(norm_src, nN);
  norm_kernel<<<(nN + 255) / 256, 256, 0, stream>>>(norm_dst, nN);

  // Pre-pack weights into WMMA B-operand layout (bf16)
  packW_kernel<<<64, 256, 0, stream>>>(W0, Wp0, 128);
  packW_kernel<<<64, 256, 0, stream>>>(W1, Wp1, 128);
  packW_kernel<<<8, 256, 0, stream>>>(W2, Wp2, 16);

  const int gemmBlocks = nN / 16;   // 3125, exact

  // Layer 0: SpMM(features * norm_src) -> GEMM(W0) + bias + relu
  zero_f32<<<25000, 256, 0, stream>>>(bufB, nN * 128);
  spmm128_kernel<<<(nE + 7) / 8, 256, 0, stream>>>(features, norm_src, src, dst, bufB, nE);
  gemm_wmma_kernel<8, true><<<gemmBlocks, 256, 0, stream>>>(bufB, norm_dst, Wp0, b0, bufA);

  // Layer 1: SpMM(bufA * norm_src) -> GEMM(W1) + bias + relu
  zero_f32<<<25000, 256, 0, stream>>>(bufB, nN * 128);
  spmm128_kernel<<<(nE + 7) / 8, 256, 0, stream>>>(bufA, norm_src, src, dst, bufB, nE);
  gemm_wmma_kernel<8, true><<<gemmBlocks, 256, 0, stream>>>(bufB, norm_dst, Wp1, b1, bufA);

  // Layer 2: transform first (128 -> 16), then SpMM, then bias + sigmoid
  gemm_wmma_kernel<1, false><<<gemmBlocks, 32, 0, stream>>>(bufA, norm_src, Wp2, nullptr, h2);
  zero_f32<<<(nN * 16 + 255) / 256, 256, 0, stream>>>(agg2, nN * 16);
  spmm16_kernel<<<(nE * 16 + 255) / 256, 256, 0, stream>>>(h2, src, dst, agg2, nE);
  final_kernel<<<(nN * 16 + 255) / 256, 256, 0, stream>>>(agg2, norm_dst, b2, out, nN);
}
